// EMSA_35768487641146
// MI455X (gfx1250) — matrix operand order repaired and checked by reading, not compiled, not run on hardware
//
#include <hip/hip_runtime.h>

// ---------------------------------------------------------------------------
// EMSA (spatial-reduction multi-head attention) for MI455X / gfx1250, wave32.
// All GEMMs use v_wmma_f32_16x16x32_bf16 (bf16 in, fp32 accumulate) with
// 2(M) x 4(N) register tiling per wave (32x64 macro-tile, 8 accumulators).
// ---------------------------------------------------------------------------

typedef __attribute__((ext_vector_type(16))) __bf16 v16bf;
typedef __attribute__((ext_vector_type(8)))  __bf16 v8bf;
typedef __attribute__((ext_vector_type(8)))  float  v8f;

union Frag { v16bf v; v8bf h[2]; };

#define B_      4
#define NQ_     3136
#define C_      512
#define HIMG_   56
#define NK_     784      // 28*28
#define NKP_    800      // padded K-dim for att@V (multiple of 32)
#define HEADS_  8
#define DM_     512
#define NHID_   4096     // HEADS * DM
#define LN_EPS_ 1e-3f
#define BN_EPS_ 1e-3f
#define QSCALE_ 0.04419417382415922f   // 1/sqrt(512)

// ---------------------------------------------------------------------------
// Wave-level 32x(16*NV) WMMA macro-tile: acc[i][j] += A[i] * B[j].
// A0: 32 rows, stride lda. B0: NV groups of 16 "columns" stored as rows of
// K contiguous elements, stride ldb. kdim must be a multiple of 32.
// ---------------------------------------------------------------------------
template <int NV>
__device__ __forceinline__ void wmma_mt(const __bf16* __restrict__ A0, int lda,
                                        const __bf16* __restrict__ B0, int ldb,
                                        int kdim, v8f acc[2][4])
{
    const int lane = threadIdx.x & 31;
    const int r    = lane & 15;
    const int hh   = lane >> 4;
    const __bf16* pa0 = A0 + (long)r * lda;
    const __bf16* pa1 = A0 + (long)(r + 16) * lda;
    const __bf16* pb  = B0 + (long)r * ldb;

    for (int k0 = 0; k0 < kdim; k0 += 32) {
        Frag a0, a1, b[NV];
        a0.h[0] = *(const v8bf*)(pa0 + k0 + 8 * hh);
        a0.h[1] = *(const v8bf*)(pa0 + k0 + 16 + 8 * hh);
        a1.h[0] = *(const v8bf*)(pa1 + k0 + 8 * hh);
        a1.h[1] = *(const v8bf*)(pa1 + k0 + 16 + 8 * hh);
#pragma unroll
        for (int j = 0; j < NV; ++j) {
            b[j].h[0] = *(const v8bf*)(pb + (long)j * 16 * ldb + k0 + 8 * hh);
            b[j].h[1] = *(const v8bf*)(pb + (long)j * 16 * ldb + k0 + 16 + 8 * hh);
        }
#pragma unroll
        for (int j = 0; j < NV; ++j) {
            acc[0][j] = __builtin_amdgcn_wmma_f32_16x16x32_bf16(
                            false, a0.v, false, b[j].v, (short)0, acc[0][j], false, false);
            acc[1][j] = __builtin_amdgcn_wmma_f32_16x16x32_bf16(
                            false, a1.v, false, b[j].v, (short)0, acc[1][j], false, false);
        }
    }
}

// ---------------------------------------------------------------------------
// Elementwise fp32 -> bf16
// ---------------------------------------------------------------------------
__global__ void cvt_bf16_kernel(const float* __restrict__ in, __bf16* __restrict__ out, long n)
{
    long i = (long)blockIdx.x * 256 + threadIdx.x;
    if (i < n) out[i] = (__bf16)in[i];
}

// W[K][N] fp32 -> WT[N][K] bf16
__global__ void transpose_cvt_kernel(const float* __restrict__ in, __bf16* __restrict__ out,
                                     int K, int N)
{
    long i = (long)blockIdx.x * 256 + threadIdx.x;
    if (i < (long)K * N) {
        int k = (int)(i / N), n = (int)(i % N);
        out[(long)n * K + k] = (__bf16)in[i];
    }
}

// Zero the padded K-region (784..799) of Vt rows.
__global__ void zeropad_vt_kernel(__bf16* __restrict__ Vt)
{
    int i   = blockIdx.x * 256 + threadIdx.x;   // 32*512*16 = 262144 total
    int row = i >> 4;
    int k   = NK_ + (i & 15);
    Vt[(long)row * NKP_ + k] = (__bf16)0.f;
}

// ---------------------------------------------------------------------------
// Depthwise 3x3 stride-2 SAME conv on queries (b,56,56,512) + LayerNorm(C).
// One block per (b, out-pixel); 256 threads x 2 channels.
// ---------------------------------------------------------------------------
__global__ void sr_ln_kernel(const float* __restrict__ q, const float* __restrict__ kern,
                             const float* __restrict__ bias, const float* __restrict__ gamma,
                             const float* __restrict__ beta, __bf16* __restrict__ xbf)
{
    const int p  = blockIdx.x;             // 0..783
    const int b  = blockIdx.y;
    const int oy = p / 28, ox = p % 28;
    const int t  = threadIdx.x;

    float vals[2];
    for (int i = 0; i < 2; ++i) {
        int c = t + i * 256;
        float acc = bias[c];
        for (int ky = 0; ky < 3; ++ky) {
            int iy = oy * 2 + ky;
            if (iy >= HIMG_) continue;
            for (int kx = 0; kx < 3; ++kx) {
                int ix = ox * 2 + kx;
                if (ix >= HIMG_) continue;
                acc += kern[(ky * 3 + kx) * C_ + c] *
                       q[((long)b * NQ_ + iy * HIMG_ + ix) * C_ + c];
            }
        }
        vals[i] = acc;
    }

    __shared__ float rs[256], rs2[256];
    rs[t]  = vals[0] + vals[1];
    rs2[t] = vals[0] * vals[0] + vals[1] * vals[1];
    __syncthreads();
    for (int off = 128; off > 0; off >>= 1) {
        if (t < off) { rs[t] += rs[t + off]; rs2[t] += rs2[t + off]; }
        __syncthreads();
    }
    float mu  = rs[0] * (1.0f / C_);
    float var = rs2[0] * (1.0f / C_) - mu * mu;
    float inv = rsqrtf(var + LN_EPS_);
    for (int i = 0; i < 2; ++i) {
        int c = t + i * 256;
        float y = (vals[i] - mu) * inv * gamma[c] + beta[c];
        xbf[((long)b * NK_ + p) * C_ + c] = (__bf16)y;
    }
}

// ---------------------------------------------------------------------------
// Projection GEMM: C[M, 4096] = A[M,512] @ W + bias, scattered per head.
// MODE 1: Q  -> Qh[((b*8+h)*3136 + r)*512 + d], scaled by 1/sqrt(512)
// MODE 2: K  -> Kh[((b*8+h)*784  + r)*512 + d]
// MODE 3: Vt -> Vt[((b*8+h)*512  + d)*800 + r]   (transposed, padded ld)
// Block = 64(M) x 256(N); wave = 32(M) x 64(N).
// ---------------------------------------------------------------------------
template <int ROWS, int MODE>
__global__ void __launch_bounds__(256)
gemm_proj_kernel(const __bf16* __restrict__ A, const __bf16* __restrict__ WT,
                 const float* __restrict__ bias, __bf16* __restrict__ out, float scale)
{
    const int wave = threadIdx.x >> 5;
    const int wm   = wave & 1, wn = wave >> 1;
    const int tm0  = blockIdx.y * 4 + wm * 2;
    const int tn0  = blockIdx.x * 16 + wn * 4;

    const __bf16* Abase = A  + (long)tm0 * 16 * C_;
    const __bf16* Bbase = WT + (long)tn0 * 16 * C_;
    v8f acc[2][4] = {};
    wmma_mt<4>(Abase, C_, Bbase, C_, C_, acc);

    const int lane = threadIdx.x & 31;
    const int r    = lane & 15;
    const int hh   = lane >> 4;
    for (int j = 0; j < 4; ++j) {
        const int n = (tn0 + j) * 16 + r;
        const int h = n >> 9;
        const int d = n & 511;
        const float bvv = bias[n];
        for (int i = 0; i < 2; ++i) {
#pragma unroll
            for (int jj = 0; jj < 8; ++jj) {
                int m  = (tm0 + i) * 16 + jj + 8 * hh;
                int b  = m / ROWS;          // compile-time divisor
                int rr = m % ROWS;
                float v = (acc[i][j][jj] + bvv) * scale;
                long idx;
                if (MODE == 3)
                    idx = ((long)(b * HEADS_ + h) * DM_ + d) * NKP_ + rr;
                else
                    idx = ((long)(b * HEADS_ + h) * ROWS + rr) * DM_ + d;
                out[idx] = (__bf16)v;
            }
        }
    }
}

// ---------------------------------------------------------------------------
// Scores: S[bh, q, k] = sum_d Qh[bh,q,d] * Kh[bh,k,d]   (scale folded into Qh)
// Block = 64(M) x 256(N); N-tiles = 49 -> wave-uniform remainder handling.
// ---------------------------------------------------------------------------
__global__ void __launch_bounds__(256)
gemm_scores_kernel(const __bf16* __restrict__ Qh, const __bf16* __restrict__ Kh,
                   float* __restrict__ S)
{
    const int bh   = blockIdx.z;
    const int wave = threadIdx.x >> 5;
    const int wm   = wave & 1, wn = wave >> 1;
    const int tm0  = blockIdx.y * 4 + wm * 2;
    const int tn0  = blockIdx.x * 16 + wn * 4;

    int nv = (NK_ / 16) - tn0;              // 49 tiles total
    if (nv <= 0) return;                    // wave-uniform exit
    if (nv > 4) nv = 4;

    const __bf16* Abase = Qh + (long)bh * NQ_ * C_ + (long)tm0 * 16 * C_;
    const __bf16* Bbase = Kh + (long)bh * NK_ * C_ + (long)tn0 * 16 * C_;
    v8f acc[2][4] = {};
    switch (nv) {
        case 4: wmma_mt<4>(Abase, C_, Bbase, C_, C_, acc); break;
        case 3: wmma_mt<3>(Abase, C_, Bbase, C_, C_, acc); break;
        case 2: wmma_mt<2>(Abase, C_, Bbase, C_, C_, acc); break;
        default: wmma_mt<1>(Abase, C_, Bbase, C_, C_, acc); break;
    }

    const int lane = threadIdx.x & 31;
    const int r    = lane & 15;
    const int hh   = lane >> 4;
    float* Sb = S + (long)bh * NQ_ * NK_;
    for (int j = 0; j < nv; ++j) {
        const int n = (tn0 + j) * 16 + r;
        for (int i = 0; i < 2; ++i) {
#pragma unroll
            for (int jj = 0; jj < 8; ++jj) {
                int m = (tm0 + i) * 16 + jj + 8 * hh;
                Sb[(long)m * NK_ + n] = acc[i][j][jj];
            }
        }
    }
}

// ---------------------------------------------------------------------------
// Head-mix (8x8 conv over h) + softmax over k + BN-stat partials.
// One block per (b, q): reads all 8 head rows of S once, produces 8 prob rows
// (bf16, padded to 800) and per-(b,q,g) partial sums for the BatchNorm stats.
// ---------------------------------------------------------------------------
__global__ void mix_softmax_kernel(const float* __restrict__ S, const float* __restrict__ tw,
                                   const float* __restrict__ tb, __bf16* __restrict__ P,
                                   float* __restrict__ partial)
{
    const int qi = blockIdx.x;
    const int b  = blockIdx.y;
    const int t  = threadIdx.x;

    __shared__ float sm[HEADS_ * NK_];   // 25 KB
    __shared__ float red1[256], red2[256];
    __shared__ float stw[64], stb[8];
    if (t < 64) stw[t] = tw[t];
    if (t < 8)  stb[t] = tb[t];
    __syncthreads();

    const long baseS = (long)b * HEADS_ * NQ_ * NK_ + (long)qi * NK_;
    for (int k = t; k < NK_; k += 256) {
        float hv[8];
        for (int h = 0; h < 8; ++h) hv[h] = S[baseS + (long)h * NQ_ * NK_ + k];
        for (int g = 0; g < 8; ++g) {
            float acc = stb[g];
            for (int h = 0; h < 8; ++h) acc += hv[h] * stw[h * 8 + g];
            sm[g * NK_ + k] = acc;
        }
    }
    __syncthreads();

    for (int g = 0; g < 8; ++g) {
        float* row = sm + g * NK_;
        float m = -3.4e38f;
        for (int k = t; k < NK_; k += 256) m = fmaxf(m, row[k]);
        red1[t] = m; __syncthreads();
        for (int off = 128; off > 0; off >>= 1) {
            if (t < off) red1[t] = fmaxf(red1[t], red1[t + off]);
            __syncthreads();
        }
        m = red1[0]; __syncthreads();

        float se = 0.f;
        for (int k = t; k < NK_; k += 256) {
            float e = __expf(row[k] - m);
            row[k] = e; se += e;
        }
        red1[t] = se; __syncthreads();
        for (int off = 128; off > 0; off >>= 1) {
            if (t < off) red1[t] += red1[t + off];
            __syncthreads();
        }
        float inv = 1.0f / red1[0]; __syncthreads();

        __bf16* prow = P + ((long)(b * HEADS_ + g) * NQ_ + qi) * NKP_;
        float ps = 0.f, ps2 = 0.f;
        for (int k = t; k < NK_; k += 256) {
            float pv = row[k] * inv;
            prow[k] = (__bf16)pv;
            ps += pv; ps2 += pv * pv;
        }
        for (int k = NK_ + t; k < NKP_; k += 256) prow[k] = (__bf16)0.f;

        red1[t] = ps; red2[t] = ps2; __syncthreads();
        for (int off = 128; off > 0; off >>= 1) {
            if (t < off) { red1[t] += red1[t + off]; red2[t] += red2[t + off]; }
            __syncthreads();
        }
        if (t == 0) {
            float* pp = partial + ((long)(b * NQ_ + qi) * HEADS_ + g) * 2;
            pp[0] = red1[0]; pp[1] = red2[0];
        }
        __syncthreads();
    }
}

// Reduce partials -> per-(b,g) BN affine coefficients a, c.
__global__ void stats_kernel(const float* __restrict__ partial, const float* __restrict__ bng,
                             const float* __restrict__ bnb, float* __restrict__ coef)
{
    const int bg = blockIdx.x;           // 0..31
    const int b = bg >> 3, g = bg & 7, t = threadIdx.x;
    float s = 0.f, s2 = 0.f;
    for (int qi = t; qi < NQ_; qi += 256) {
        const float* pp = partial + ((long)(b * NQ_ + qi) * HEADS_ + g) * 2;
        s += pp[0]; s2 += pp[1];
    }
    __shared__ float r1[256], r2[256];
    r1[t] = s; r2[t] = s2; __syncthreads();
    for (int off = 128; off > 0; off >>= 1) {
        if (t < off) { r1[t] += r1[t + off]; r2[t] += r2[t + off]; }
        __syncthreads();
    }
    if (t == 0) {
        const float N = (float)NQ_ * (float)NK_;
        float mean = r1[0] / N;
        float var  = r2[0] / N - mean * mean;
        float a = bng[b * HEADS_ + g] * rsqrtf(var + BN_EPS_);
        coef[bg * 2 + 0] = a;
        coef[bg * 2 + 1] = bnb[b * HEADS_ + g] - a * mean;
    }
}

// Vsum[bh, d] = sum_k Vt[bh, d, k]
__global__ void vsum_kernel(const __bf16* __restrict__ Vt, float* __restrict__ Vsum)
{
    const int bh = blockIdx.x, t = threadIdx.x;
    for (int d = t; d < DM_; d += 256) {
        const __bf16* p = Vt + ((long)bh * DM_ + d) * NKP_;
        float s = 0.f;
        for (int k = 0; k < NK_; ++k) s += (float)p[k];
        Vsum[bh * DM_ + d] = s;
    }
}

// ---------------------------------------------------------------------------
// att@V with fused BatchNorm: O2[b,q, g*512+d] = a*(P@V)[q,d] + c*Vsum[d]
// Block = 64(M) x 256(N); N = 512 -> grid.x = 2.
// ---------------------------------------------------------------------------
__global__ void __launch_bounds__(256)
gemm_pv_kernel(const __bf16* __restrict__ P, const __bf16* __restrict__ Vt,
               const float* __restrict__ coef, const float* __restrict__ Vsum,
               __bf16* __restrict__ O2)
{
    const int bh   = blockIdx.z;
    const int wave = threadIdx.x >> 5;
    const int wm   = wave & 1, wn = wave >> 1;
    const int tm0  = blockIdx.y * 4 + wm * 2;
    const int tn0  = blockIdx.x * 16 + wn * 4;

    const __bf16* Abase = P  + (long)bh * NQ_ * NKP_ + (long)tm0 * 16 * NKP_;
    const __bf16* Bbase = Vt + (long)bh * DM_ * NKP_ + (long)tn0 * 16 * NKP_;
    v8f acc[2][4] = {};
    wmma_mt<4>(Abase, NKP_, Bbase, NKP_, NKP_, acc);

    const float a = coef[bh * 2 + 0];
    const float c = coef[bh * 2 + 1];
    const int lane = threadIdx.x & 31;
    const int r    = lane & 15;
    const int hh   = lane >> 4;
    const int b = bh >> 3, g = bh & 7;

    for (int j = 0; j < 4; ++j) {
        const int d  = (tn0 + j) * 16 + r;
        const float vs = Vsum[bh * DM_ + d];
        for (int i = 0; i < 2; ++i) {
#pragma unroll
            for (int jj = 0; jj < 8; ++jj) {
                int q = (tm0 + i) * 16 + jj + 8 * hh;
                O2[((long)(b * NQ_ + q)) * NHID_ + g * DM_ + d] =
                    (__bf16)(a * acc[i][j][jj] + c * vs);
            }
        }
    }
}

// Final projection: out[12544, 512] = O2[12544, 4096] @ Wo + bo  (fp32 out)
__global__ void __launch_bounds__(256)
gemm_out_kernel(const __bf16* __restrict__ O2, const __bf16* __restrict__ WoT,
                const float* __restrict__ bo, float* __restrict__ out)
{
    const int wave = threadIdx.x >> 5;
    const int wm   = wave & 1, wn = wave >> 1;
    const int tm0  = blockIdx.y * 4 + wm * 2;
    const int tn0  = blockIdx.x * 16 + wn * 4;

    const __bf16* Abase = O2  + (long)tm0 * 16 * NHID_;
    const __bf16* Bbase = WoT + (long)tn0 * 16 * NHID_;
    v8f acc[2][4] = {};
    wmma_mt<4>(Abase, NHID_, Bbase, NHID_, NHID_, acc);

    const int lane = threadIdx.x & 31;
    const int r    = lane & 15;
    const int hh   = lane >> 4;
    for (int j = 0; j < 4; ++j) {
        const int n = (tn0 + j) * 16 + r;
        const float bvv = bo[n];
        for (int i = 0; i < 2; ++i) {
#pragma unroll
            for (int jj = 0; jj < 8; ++jj) {
                int m = (tm0 + i) * 16 + jj + 8 * hh;
                out[(long)m * C_ + n] = acc[i][j][jj] + bvv;
            }
        }
    }
}

// ---------------------------------------------------------------------------
extern "C" void kernel_launch(void* const* d_in, const int* in_sizes, int n_in,
                              void* d_out, int out_size, void* d_ws, size_t ws_size,
                              hipStream_t stream)
{
    const float* queries = (const float*)d_in[0];
    const float* Wq  = (const float*)d_in[3];
    const float* bq  = (const float*)d_in[4];
    const float* Wk  = (const float*)d_in[5];
    const float* bk  = (const float*)d_in[6];
    const float* Wv  = (const float*)d_in[7];
    const float* bv  = (const float*)d_in[8];
    const float* Wo  = (const float*)d_in[9];
    const float* bo  = (const float*)d_in[10];
    const float* srk = (const float*)d_in[11];
    const float* srb = (const float*)d_in[12];
    const float* lng = (const float*)d_in[13];
    const float* lnb = (const float*)d_in[14];
    const float* tw  = (const float*)d_in[15];
    const float* tb  = (const float*)d_in[16];
    const float* bng = (const float*)d_in[17];
    const float* bnb = (const float*)d_in[18];
    float* out = (float*)d_out;

    char* ws = (char*)d_ws;
    auto alloc = [&](size_t bytes) -> char* {
        char* p = ws;
        ws += (bytes + 255) & ~(size_t)255;
        return p;
    };

    __bf16* qbf  = (__bf16*)alloc((size_t)B_ * NQ_ * C_ * 2);          // 12.8 MB
    __bf16* xbf  = (__bf16*)alloc((size_t)B_ * NK_ * C_ * 2);          //  3.2 MB
    __bf16* WqT  = (__bf16*)alloc((size_t)NHID_ * C_ * 2);             //  4.2 MB
    __bf16* WkT  = (__bf16*)alloc((size_t)NHID_ * C_ * 2);
    __bf16* WvT  = (__bf16*)alloc((size_t)NHID_ * C_ * 2);
    __bf16* WoT  = (__bf16*)alloc((size_t)C_ * NHID_ * 2);
    __bf16* Qh   = (__bf16*)alloc((size_t)B_ * HEADS_ * NQ_ * DM_ * 2);   // 102.8 MB
    __bf16* Kh   = (__bf16*)alloc((size_t)B_ * HEADS_ * NK_ * DM_ * 2);   //  25.7 MB
    __bf16* Vt   = (__bf16*)alloc((size_t)B_ * HEADS_ * DM_ * NKP_ * 2);  //  26.2 MB
    float*  S    = (float*) alloc((size_t)B_ * HEADS_ * NQ_ * NK_ * 4);   // 314.7 MB
    __bf16* P    = (__bf16*)alloc((size_t)B_ * HEADS_ * NQ_ * NKP_ * 2);  // 160.6 MB
    __bf16* O2   = (__bf16*)alloc((size_t)B_ * NQ_ * NHID_ * 2);          // 102.8 MB
    float*  part = (float*) alloc((size_t)B_ * NQ_ * HEADS_ * 2 * 4);     //   0.8 MB
    float*  coef = (float*) alloc((size_t)B_ * HEADS_ * 2 * 4);
    float*  Vsum = (float*) alloc((size_t)B_ * HEADS_ * DM_ * 4);

    // 1) bf16 conversions
    {
        long n = (long)B_ * NQ_ * C_;
        cvt_bf16_kernel<<<(int)((n + 255) / 256), 256, 0, stream>>>(queries, qbf, n);
    }
    transpose_cvt_kernel<<<(C_ * NHID_ + 255) / 256, 256, 0, stream>>>(Wq, WqT, C_, NHID_);
    transpose_cvt_kernel<<<(C_ * NHID_ + 255) / 256, 256, 0, stream>>>(Wk, WkT, C_, NHID_);
    transpose_cvt_kernel<<<(C_ * NHID_ + 255) / 256, 256, 0, stream>>>(Wv, WvT, C_, NHID_);
    transpose_cvt_kernel<<<(NHID_ * C_ + 255) / 256, 256, 0, stream>>>(Wo, WoT, NHID_, C_);

    // 2) spatial reduction conv + LayerNorm -> xbf
    sr_ln_kernel<<<dim3(NK_, B_), 256, 0, stream>>>(queries, srk, srb, lng, lnb, xbf);

    // 3) projections (WMMA). Block covers 64 rows x 256 cols.
    gemm_proj_kernel<NQ_, 1><<<dim3(16, (B_ * NQ_) / 64), 256, 0, stream>>>(
        qbf, WqT, bq, Qh, QSCALE_);
    gemm_proj_kernel<NK_, 2><<<dim3(16, (B_ * NK_) / 64), 256, 0, stream>>>(
        xbf, WkT, bk, Kh, 1.0f);
    gemm_proj_kernel<NK_, 3><<<dim3(16, (B_ * NK_) / 64), 256, 0, stream>>>(
        xbf, WvT, bv, Vt, 1.0f);
    zeropad_vt_kernel<<<(B_ * HEADS_ * DM_ * 16) / 256, 256, 0, stream>>>(Vt);

    // 4) scores S = Q K^T / sqrt(dk)  (WMMA, scale folded into Qh)
    //    49 N-tiles -> grid.x = 4 blocks of 16 tiles (remainder wave-uniform).
    gemm_scores_kernel<<<dim3(4, NQ_ / 64, B_ * HEADS_), 256, 0, stream>>>(Qh, Kh, S);

    // 5) head-mix + softmax + BN partial stats
    mix_softmax_kernel<<<dim3(NQ_, B_), 256, 0, stream>>>(S, tw, tb, P, part);

    // 6) BN coefficients + V column sums (BN folded into att@V epilogue)
    stats_kernel<<<B_ * HEADS_, 256, 0, stream>>>(part, bng, bnb, coef);
    vsum_kernel<<<B_ * HEADS_, 256, 0, stream>>>(Vt, Vsum);

    // 7) att@V with fused BN (WMMA)
    gemm_pv_kernel<<<dim3(2, NQ_ / 64, B_ * HEADS_), 256, 0, stream>>>(P, Vt, coef, Vsum, O2);

    // 8) output projection (WMMA, fp32 out)
    gemm_out_kernel<<<dim3(2, (B_ * NQ_) / 64), 256, 0, stream>>>(O2, WoT, bo, out);

    (void)in_sizes; (void)n_in; (void)out_size; (void)ws_size;
}
